// MultiHeadAttention_2602750181855
// MI455X (gfx1250) — matrix operand order repaired and checked by reading, not compiled, not run on hardware
//
#include <hip/hip_runtime.h>
#include <cstdint>

typedef __bf16 v16bf __attribute__((ext_vector_type(16)));
typedef float  v8f   __attribute__((ext_vector_type(8)));

#define HIDDEN   1024
#define SEQ      2048
#define BATCH    2
#define HEADS    16
#define HEAD_DIM 64

__device__ __forceinline__ unsigned short f2bf(float f){
  union { float f; unsigned u; } x; x.f = f;
  unsigned r = x.u + 0x7FFFu + ((x.u >> 16) & 1u);   // round-to-nearest-even
  return (unsigned short)(r >> 16);
}

union FragAB { v16bf v; uint4 q[2]; unsigned short s[16]; };

// CDNA5 async memory->LDS copy (ASYNCcnt-tracked, no VGPR data path).
__device__ __forceinline__ void async_load_b128(const void* gaddr, void* lds_ptr){
  unsigned lds = (unsigned)(uintptr_t)lds_ptr;                 // low 32 bits = LDS offset
  unsigned long long ga = (unsigned long long)(uintptr_t)gaddr;
  asm volatile("global_load_async_to_lds_b128 %0, %1, off"
               :: "v"(lds), "v"(ga) : "memory");
}

// ---------------- conversion kernels (one-time, tiny vs 537MB attn write) ---
__global__ void cvt_f32_to_bf16(const float* __restrict__ in,
                                unsigned short* __restrict__ out, int n){
  for (int i = blockIdx.x*blockDim.x + threadIdx.x; i < n; i += gridDim.x*blockDim.x)
    out[i] = f2bf(in[i]);
}

// Wt[n][k] = W[k][n]  (so WMMA B-fragments read K-contiguous 8-half chunks per lane)
__global__ void cvt_transpose_bf16(const float* __restrict__ W,
                                   unsigned short* __restrict__ Wt, int K, int N){
  int total = K * N;
  for (int i = blockIdx.x*blockDim.x + threadIdx.x; i < total; i += gridDim.x*blockDim.x){
    int k = i / N, n = i - k * N;
    Wt[(size_t)n*K + k] = f2bf(W[i]);
  }
}

// ---------------- GEMM: out[M,N] = A[M,K] @ Bt[N,K]^T + bias --------------
// Double-buffered async-to-LDS staging. 128x64 block tile, 8 waves.
// OUT_MODE 0: bf16 row-major   1: bf16 scattered to [B,H,Dh,S] (V transposed)
// OUT_MODE 2: f32 row-major (final projection into d_out)
template<int OUT_MODE>
__global__ __launch_bounds__(256)
void gemm_bias_wmma(const unsigned short* __restrict__ A,
                    const unsigned short* __restrict__ Bt,
                    const float* __restrict__ bias,
                    void* __restrict__ out,
                    int M, int N, int K){
  __shared__ unsigned short sA[2][128*32];   // 2 x 8 KB
  __shared__ unsigned short sB[2][64*32];    // 2 x 4 KB
  const int tid  = threadIdx.x;
  const int wave = tid >> 5, lane = tid & 31;
  const int ln = lane & 15, g = lane >> 4;
  const int mBase = blockIdx.y * 128, nBase = blockIdx.x * 64;

  // per-thread async staging: 2 x b128 for the A tile, 1 x b128 for the B tile
  auto issue_tile = [&](int buf, int kb){
#pragma unroll
    for (int j = 0; j < 2; j++){
      int ci = j*256 + tid;
      int r = ci >> 2, c = ci & 3;
      async_load_b128(A + (size_t)(mBase + r)*K + kb + c*8, &sA[buf][ci*8]);
    }
    {
      int n = tid >> 2, c = tid & 3;
      async_load_b128(Bt + (size_t)(nBase + n)*K + kb + c*8, &sB[buf][tid*8]);
    }
  };

  v8f acc[4]; acc[0] = {}; acc[1] = {}; acc[2] = {}; acc[3] = {};

  issue_tile(0, 0);
  for (int kb = 0, it = 0; kb < K; kb += 32, ++it){
    const int buf = it & 1;
    __syncthreads();                       // prior compute on buf^1 finished everywhere
    if (kb + 32 < K){
      issue_tile(buf ^ 1, kb + 32);        // prefetch next tile while we compute this one
      asm volatile("s_wait_asynccnt 0x3" ::: "memory");  // retire current tile's 3 ops
    } else {
      asm volatile("s_wait_asynccnt 0x0" ::: "memory");
    }
    __syncthreads();                       // current tile visible to all waves

    // A fragment: lane row = wave*16+ln; chunks [8g,8g+8) and [16+8g,+8)
    FragAB af;
    {
      const uint4* pa = (const uint4*)(&sA[buf][(wave*16 + ln)*32]);
      af.q[0] = pa[g]; af.q[1] = pa[2 + g];
    }
#pragma unroll
    for (int t = 0; t < 4; t++){
      FragAB bf;  // B fragment: lane col = t*16+ln; chunks [8g,8g+8) and [16+8g,+8)
      const uint4* pb = (const uint4*)(&sB[buf][(t*16 + ln)*32]);
      bf.q[0] = pb[g]; bf.q[1] = pb[2 + g];
      acc[t] = __builtin_amdgcn_wmma_f32_16x16x32_bf16(
                 false, af.v, false, bf.v, (short)0, acc[t], false, false);
    }
  }

#pragma unroll
  for (int t = 0; t < 4; t++){
    int col = nBase + t*16 + ln;
    float bv = bias[col];
#pragma unroll
    for (int i = 0; i < 8; i++){
      int row = mBase + wave*16 + 8*g + i;   // C/D layout: m = 8*(lane>>4)+i
      float val = acc[t][i] + bv;
      if (OUT_MODE == 0){
        ((unsigned short*)out)[(size_t)row*N + col] = f2bf(val);
      } else if (OUT_MODE == 1){
        int b = row >> 11, s = row & 2047, h = col >> 6, hd = col & 63;
        ((unsigned short*)out)[(size_t)((b*HEADS + h)*HEAD_DIM + hd)*SEQ + s] = f2bf(val);
      } else {
        ((float*)out)[(size_t)row*N + col] = val;
      }
    }
  }
}

// ---------------- fused attention: scores -> softmax -> attn out + ctx ----
// One block per (b, h, 16-query band). 8 waves. 16x2048 f32 scores in LDS.
__global__ __launch_bounds__(256)
void mha_softmax_ctx(const unsigned short* __restrict__ Q,   // [B*S,1024] bf16
                     const unsigned short* __restrict__ Km,  // [B*S,1024] bf16
                     const unsigned short* __restrict__ Vt,  // [B,H,Dh,S] bf16
                     float* __restrict__ attn_out,           // [B,H,S,S] f32
                     unsigned short* __restrict__ ctx){      // [B*S,1024] bf16
  extern __shared__ float smem[];
  float* s_scores = smem;               // 16*2048 f32 = 128 KB
  float* s_red    = smem + 16*SEQ;      // 16*16 + 16

  const int tid = threadIdx.x;
  const int wave = tid >> 5, lane = tid & 31, ln = lane & 15, g = lane >> 4;
  const int bid  = blockIdx.x;
  const int band = bid & 127;
  const int h    = (bid >> 7) & 15;
  const int b    = bid >> 11;
  const int qBase = band * 16;
  const float scale = 0.125f;           // 1/sqrt(64)

  // Q band A-fragments (shared by all score tiles of this wave), 2 K-steps
  FragAB aq[2];
#pragma unroll
  for (int ks = 0; ks < 2; ks++){
    const uint4* p = (const uint4*)(Q + (size_t)(b*SEQ + qBase + ln)*HIDDEN
                                      + h*HEAD_DIM + ks*32);
    aq[ks].q[0] = p[g]; aq[ks].q[1] = p[2 + g];
  }

  // ---- Phase 1: scores = scale * Q K^T, wave covers keys [wave*256,+256) --
  // Software-pipelined: fetch key-tile t+1's B fragments before WMMA of tile t.
  auto load_kfrag = [&](FragAB* bf, int keyBase){
    const uint4* pk = (const uint4*)(Km + (size_t)(b*SEQ + keyBase + ln)*HIDDEN
                                        + h*HEAD_DIM);
#pragma unroll
    for (int ks = 0; ks < 2; ks++){
      bf[ks].q[0] = pk[4*ks + g];
      bf[ks].q[1] = pk[4*ks + 2 + g];
    }
  };

  FragAB bcur[2], bnext[2];
  load_kfrag(bcur, wave*256);
  for (int t = 0; t < 16; t++){
    int keyBase = wave*256 + t*16;
    if (t < 15) load_kfrag(bnext, keyBase + 16);  // overlap fetch with WMMA
    v8f acc = {};
#pragma unroll
    for (int ks = 0; ks < 2; ks++)
      acc = __builtin_amdgcn_wmma_f32_16x16x32_bf16(
              false, aq[ks].v, false, bcur[ks].v, (short)0, acc, false, false);
#pragma unroll
    for (int i = 0; i < 8; i++)
      s_scores[(8*g + i)*SEQ + keyBase + ln] = acc[i] * scale;
    bcur[0] = bnext[0]; bcur[1] = bnext[1];
  }
  __syncthreads();

  // ---- Phase 2: softmax over each of the 16 rows (16 threads per row) ----
  {
    int r = tid >> 4, seg = tid & 15;
    float* row = s_scores + r*SEQ;
    int c0 = seg * 128;
    float mx = -3.0e38f;
    for (int j = 0; j < 128; j++) mx = fmaxf(mx, row[c0 + j]);
    s_red[r*16 + seg] = mx;
    __syncthreads();
    if (tid < 16){
      float m2 = -3.0e38f;
      for (int j = 0; j < 16; j++) m2 = fmaxf(m2, s_red[tid*16 + j]);
      s_red[256 + tid] = m2;
    }
    __syncthreads();
    float rm = s_red[256 + r];
    float sum = 0.f;
    for (int j = 0; j < 128; j++){
      float e = __expf(row[c0 + j] - rm);
      row[c0 + j] = e; sum += e;
    }
    s_red[r*16 + seg] = sum;
    __syncthreads();
    if (tid < 16){
      float s2 = 0.f;
      for (int j = 0; j < 16; j++) s2 += s_red[tid*16 + j];
      s_red[256 + tid] = 1.0f / s2;
    }
    __syncthreads();
    float inv = s_red[256 + r];
    float4* lrow = (float4*)(row + c0);
    float4* grow = (float4*)(attn_out
                     + (size_t)((b*HEADS + h)*SEQ + qBase + r)*SEQ + c0);
    for (int j = 0; j < 32; j++){
      float4 v = lrow[j];
      v.x *= inv; v.y *= inv; v.z *= inv; v.w *= inv;
      lrow[j] = v; grow[j] = v;        // the one mandatory 537MB write
    }
  }
  __syncthreads();

  // ---- Phase 3: ctx = attn @ V, wave accumulates keys [wave*256,+256) ----
  v8f cacc[4]; cacc[0] = {}; cacc[1] = {}; cacc[2] = {}; cacc[3] = {};
  for (int kk = 0; kk < 8; kk++){
    int key0 = wave*256 + kk*32;
    FragAB af;  // A fragment from LDS scores, cvt f32->bf16
    const float4* pr = (const float4*)(s_scores + ln*SEQ + key0);
    float4 f0 = pr[4*g], f1 = pr[4*g + 1], f2 = pr[4*g + 2], f3 = pr[4*g + 3];
    af.s[0]=f2bf(f0.x);  af.s[1]=f2bf(f0.y);  af.s[2]=f2bf(f0.z);  af.s[3]=f2bf(f0.w);
    af.s[4]=f2bf(f1.x);  af.s[5]=f2bf(f1.y);  af.s[6]=f2bf(f1.z);  af.s[7]=f2bf(f1.w);
    af.s[8]=f2bf(f2.x);  af.s[9]=f2bf(f2.y);  af.s[10]=f2bf(f2.z); af.s[11]=f2bf(f2.w);
    af.s[12]=f2bf(f3.x); af.s[13]=f2bf(f3.y); af.s[14]=f2bf(f3.z); af.s[15]=f2bf(f3.w);
#pragma unroll
    for (int t = 0; t < 4; t++){
      int hd = t*16 + ln;
      FragAB bf;  // V stored transposed -> 16 contiguous keys per lane
      const uint4* pv = (const uint4*)(Vt + (size_t)((b*HEADS + h)*HEAD_DIM + hd)*SEQ
                                          + key0 + 16*g);
      bf.q[0] = pv[0]; bf.q[1] = pv[1];
      cacc[t] = __builtin_amdgcn_wmma_f32_16x16x32_bf16(
                  false, af.v, false, bf.v, (short)0, cacc[t], false, false);
    }
  }
  __syncthreads();                       // done reading s_scores; reuse as scratch
#pragma unroll
  for (int t = 0; t < 4; t++)
#pragma unroll
    for (int i = 0; i < 8; i++)
      s_scores[wave*1024 + lane*32 + t*8 + i] = cacc[t][i];
  __syncthreads();
  for (int j = 0; j < 4; j++){           // cross-wave reduction + ctx store
    int pos = tid + j*256;
    float sum = 0.f;
#pragma unroll
    for (int w = 0; w < 8; w++) sum += s_scores[w*1024 + pos];
    int l2 = pos >> 5, t = (pos >> 3) & 3, i = pos & 7;
    int m  = 8*(l2 >> 4) + i;
    int hd = t*16 + (l2 & 15);
    ctx[(size_t)(b*SEQ + qBase + m)*HIDDEN + h*HEAD_DIM + hd] = f2bf(sum);
  }
}

// --------------------------------------------------------------------------
extern "C" void kernel_launch(void* const* d_in, const int* in_sizes, int n_in,
                              void* d_out, int out_size, void* d_ws, size_t ws_size,
                              hipStream_t stream){
  const float* x  = (const float*)d_in[0];
  const float* Wq = (const float*)d_in[1];
  const float* bq = (const float*)d_in[2];
  const float* Wk = (const float*)d_in[3];
  const float* bk = (const float*)d_in[4];
  const float* Wv = (const float*)d_in[5];
  const float* bv = (const float*)d_in[6];
  const float* Wo = (const float*)d_in[7];
  const float* bo = (const float*)d_in[8];

  const size_t MB = (size_t)1 << 20;
  char* ws = (char*)d_ws;                      // 48 MB used
  unsigned short* xb  = (unsigned short*)(ws +  0*MB);  // x bf16        8 MB
  unsigned short* wqt = (unsigned short*)(ws +  8*MB);  // Wq^T bf16     2 MB
  unsigned short* wkt = (unsigned short*)(ws + 10*MB);
  unsigned short* wvt = (unsigned short*)(ws + 12*MB);
  unsigned short* wot = (unsigned short*)(ws + 14*MB);
  unsigned short* Qb  = (unsigned short*)(ws + 16*MB);  // Q bf16        8 MB
  unsigned short* Kb  = (unsigned short*)(ws + 24*MB);  // K bf16        8 MB
  unsigned short* Vt  = (unsigned short*)(ws + 32*MB);  // V^T bf16      8 MB
  unsigned short* ctx = (unsigned short*)(ws + 40*MB);  // context bf16  8 MB

  const int M = BATCH * SEQ;                   // 4096

  cvt_f32_to_bf16 <<<1024, 256, 0, stream>>>(x,  xb,  M*HIDDEN);
  cvt_transpose_bf16<<<1024, 256, 0, stream>>>(Wq, wqt, HIDDEN, HIDDEN);
  cvt_transpose_bf16<<<1024, 256, 0, stream>>>(Wk, wkt, HIDDEN, HIDDEN);
  cvt_transpose_bf16<<<1024, 256, 0, stream>>>(Wv, wvt, HIDDEN, HIDDEN);
  cvt_transpose_bf16<<<1024, 256, 0, stream>>>(Wo, wot, HIDDEN, HIDDEN);

  dim3 grid(HIDDEN/64, M/128);                 // 16 x 32 blocks
  gemm_bias_wmma<0><<<grid, 256, 0, stream>>>(xb, wqt, bq, Qb, M, HIDDEN, HIDDEN);
  gemm_bias_wmma<0><<<grid, 256, 0, stream>>>(xb, wkt, bk, Kb, M, HIDDEN, HIDDEN);
  gemm_bias_wmma<1><<<grid, 256, 0, stream>>>(xb, wvt, bv, Vt, M, HIDDEN, HIDDEN);

  float* out  = (float*)d_out;
  float* attn = out + (size_t)M*HIDDEN;        // outputs concatenated flat
  size_t smem = (size_t)(16*SEQ + 16*16 + 16) * sizeof(float);  // ~132 KB LDS
  mha_softmax_ctx<<<BATCH*HEADS*(SEQ/16), 256, smem, stream>>>(Qb, Kb, Vt, attn, ctx);

  gemm_bias_wmma<2><<<grid, 256, 0, stream>>>(ctx, wot, bo, out, M, HIDDEN, HIDDEN);
}
